// MultiHeadedAttention_8143257993440
// MI455X (gfx1250) — matrix operand order repaired and checked by reading, not compiled, not run on hardware
//
#include <hip/hip_runtime.h>
#include <hip/hip_bf16.h>
#include <stdint.h>

// ---------------------------------------------------------------------------
// MI455X (gfx1250, wave32) fused paged-KV attention layer.
// Pipeline: f32->bf16 convert, paged KV gather, QKV GEMM (WMMA bf16),
// flash attention (WMMA bf16, online softmax), output GEMM (WMMA bf16).
// Round 3: fix async builtin pointer types — clang reports the builtin wants
// (global int4*, shared int4*, i32 offset, i32 cpol).
// ---------------------------------------------------------------------------

typedef __bf16 bf16_t;
typedef __attribute__((ext_vector_type(4)))  __bf16 v4bf;
typedef __attribute__((ext_vector_type(8)))  __bf16 v8bf;
typedef __attribute__((ext_vector_type(16))) __bf16 v16bf;
typedef __attribute__((ext_vector_type(8)))  float  v8f;
typedef __attribute__((ext_vector_type(4)))  int    v4i;

constexpr int kT      = 1024;   // CHUNK
constexpr int kF      = 1024;   // N_FEAT
constexpr int kH      = 16;     // N_HEAD
constexpr int kD      = 64;     // D_K
constexpr int kOffset = 3072;   // seqlen_offset (fixed by setup_inputs)
constexpr int kTotal  = 4096;   // kOffset + kT
constexpr int kBS     = 32;     // BLOCK_SIZE

// ---------------------------------------------------------------------------
// Async global->LDS staging (CDNA5 GLOBAL_LOAD_ASYNC_TO_LDS_B128, ASYNCcnt).
// Builtin signature (from clang diagnostic): arg0 = global int4*, and by the
// same convention arg1 = shared int4*, then i32 offset, i32 cpol.
// ---------------------------------------------------------------------------
#if defined(__gfx1250__) && __has_builtin(__builtin_amdgcn_global_load_async_to_lds_b128)
#define ATHENA_ASYNC_LDS 1
typedef __attribute__((address_space(1))) v4i* gv4i_ptr;  // global int4*
typedef __attribute__((address_space(3))) v4i* lv4i_ptr;  // shared int4*
__device__ __forceinline__ void async_copy16(void* lds, const void* g) {
  __builtin_amdgcn_global_load_async_to_lds_b128(
      (gv4i_ptr)(void*)g, (lv4i_ptr)lds, 0, 0);
}
__device__ __forceinline__ void async_wait0() {
#if __has_builtin(__builtin_amdgcn_s_wait_asynccnt)
  __builtin_amdgcn_s_wait_asynccnt(0);
#else
  asm volatile("s_wait_asynccnt 0x0" ::: "memory");
#endif
}
#else
#define ATHENA_ASYNC_LDS 0
__device__ __forceinline__ void async_copy16(void* lds, const void* g) {
  *(v8bf*)lds = *(const v8bf*)g;  // 16B generic copy -> ds_store_b128
}
__device__ __forceinline__ void async_wait0() {}
#endif

__device__ __forceinline__ v8f wmma_bf16(v16bf a, v16bf b, v8f c) {
  // D = A(16x32 bf16) x B(32x16 bf16) + C(16x16 f32)
  return __builtin_amdgcn_wmma_f32_16x16x32_bf16(false, a, false, b,
                                                 (short)0, c, false, false);
}

__device__ __forceinline__ v16bf cat8(v8bf lo, v8bf hi) {
  return __builtin_shufflevector(lo, hi, 0, 1, 2, 3, 4, 5, 6, 7,
                                 8, 9, 10, 11, 12, 13, 14, 15);
}

// A-fragment (16x32 bf16) from row-major [M][K] storage, row stride ld.
// ISA 7.12.2: lanes 0-15 hold K={0..7,16..23}, lanes 16-31 K={8..15,24..31}.
__device__ __forceinline__ v16bf load_a_frag(const bf16_t* base, int ld,
                                             int l15, int hh) {
  const bf16_t* p = base + l15 * ld + hh * 8;
  v8bf lo = *(const v8bf*)p;
  v8bf hi = *(const v8bf*)(p + 16);
  return cat8(lo, hi);
}

// B-fragment (32x16 bf16) from [N][K]-transposed storage, row stride ld.
// Lane (hh,l15) holds N=nrow, K = kbase + {hh*8+0..7, 16+hh*8+0..7}.
__device__ __forceinline__ v16bf load_b_frag(const bf16_t* base, int ld,
                                             int nrow, int kbase, int hh) {
  const bf16_t* p = base + nrow * ld + kbase + hh * 8;
  v8bf lo = *(const v8bf*)p;
  v8bf hi = *(const v8bf*)(p + 16);
  return cat8(lo, hi);
}

// ---------------------------------------------------------------------------
// Kernel 1: f32 -> bf16 conversion (vectorized x4).
// ---------------------------------------------------------------------------
__global__ void __launch_bounds__(256)
cvt_f32_bf16_kernel(const float* __restrict__ src, bf16_t* __restrict__ dst,
                    int n4) {
  int i = blockIdx.x * 256 + threadIdx.x;
  if (i >= n4) return;
  float4 v = *(const float4*)(src + (size_t)i * 4);
  v4bf o;
  o[0] = (bf16_t)v.x; o[1] = (bf16_t)v.y;
  o[2] = (bf16_t)v.z; o[3] = (bf16_t)v.w;
  *(v4bf*)(dst + (size_t)i * 4) = o;
}

// ---------------------------------------------------------------------------
// Kernel 2: gather historical K/V (pos < 3072) through the block table into
// contiguous bf16 [head][pos][d] tensors. New rows (pos >= 3072) are written
// by the QKV GEMM epilogue.
// ---------------------------------------------------------------------------
__global__ void __launch_bounds__(256)
gather_kv_kernel(const float* __restrict__ kc, const float* __restrict__ vc,
                 const int* __restrict__ btab, bf16_t* __restrict__ Kf,
                 bf16_t* __restrict__ Vf) {
  int idx = blockIdx.x * 256 + threadIdx.x;  // [0, 3072*16*16)
  int d4  = (idx & 15) * 4;
  int h   = (idx >> 4) & 15;
  int pos = idx >> 8;
  int phys = btab[pos >> 5];
  int so   = pos & (kBS - 1);
  size_t src = (((size_t)phys * kBS + so) * kH + h) * kD + d4;
  size_t dst = ((size_t)h * kTotal + pos) * kD + d4;
  float4 kv = *(const float4*)(kc + src);
  float4 vv = *(const float4*)(vc + src);
  v4bf ko, vo;
  ko[0] = (bf16_t)kv.x; ko[1] = (bf16_t)kv.y;
  ko[2] = (bf16_t)kv.z; ko[3] = (bf16_t)kv.w;
  vo[0] = (bf16_t)vv.x; vo[1] = (bf16_t)vv.y;
  vo[2] = (bf16_t)vv.z; vo[3] = (bf16_t)vv.w;
  *(v4bf*)(Kf + dst) = ko;
  *(v4bf*)(Vf + dst) = vo;
}

// ---------------------------------------------------------------------------
// Kernel 3/5: 1024x1024x1024 GEMM, 128x128 macro-tile, 8 waves, each wave a
// 32x64 tile (2x4 WMMA frags), K-steps of 32, double-buffered LDS.
// qkv=1: grid.z selects Q/K/V, epilogue scatters bf16 into attention layouts.
// qkv=0: f32 out + bias.
// ---------------------------------------------------------------------------
struct GemmArgs {
  const bf16_t* X;
  const bf16_t* W0; const bf16_t* W1; const bf16_t* W2;
  const float*  b0; const float*  b1; const float*  b2;
  bf16_t* dq; bf16_t* dk; bf16_t* dv;
  float*  fout;
  int qkv;
};

__global__ void __launch_bounds__(256) gemm_wmma_kernel(GemmArgs args) {
  constexpr int LDA = 40, LDB = 40;  // bf16 elements, padded vs bank conflicts
  __shared__ alignas(16) bf16_t As[2][128 * LDA];  // [M][K] row-major
  __shared__ alignas(16) bf16_t Bs[2][128 * LDB];  // [N][K] transposed

  int tid  = threadIdx.x;
  int wave = tid >> 5, lane = tid & 31;
  int hh = lane >> 4, l15 = lane & 15;
  int wm = (wave & 3) * 32;   // wave row offset in macro tile
  int wn = (wave >> 2) * 64;  // wave col offset in macro tile
  int m0 = blockIdx.y * 128, n0 = blockIdx.x * 128;

  int sel = args.qkv ? (int)blockIdx.z : 0;
  const bf16_t* W    = (sel == 0) ? args.W0 : (sel == 1) ? args.W1 : args.W2;
  const float*  bias = (sel == 0) ? args.b0 : (sel == 1) ? args.b1 : args.b2;

  // Stage A tile 128x32 as 16B async copies (512 chunks of 8 bf16).
  auto stage_a = [&](int buf, int k0) {
#pragma unroll
    for (int i = 0; i < 2; i++) {
      int c = tid + 256 * i;
      int r = c >> 2, cc = (c & 3) * 8;
      async_copy16(&As[buf][r * LDA + cc],
                   args.X + (size_t)(m0 + r) * kF + k0 + cc);
    }
  };
  // Stage B tile 32x128 transposed into [N][K] (VGPR path, scatter stores).
  auto stage_b = [&](int buf, int k0) {
#pragma unroll
    for (int i = 0; i < 4; i++) {
      int c = tid + 256 * i;
      int kk = c >> 5, nn = (c & 31) * 4;
      v4bf t = *(const v4bf*)(W + (size_t)(k0 + kk) * kF + n0 + nn);
      Bs[buf][(nn + 0) * LDB + kk] = t[0];
      Bs[buf][(nn + 1) * LDB + kk] = t[1];
      Bs[buf][(nn + 2) * LDB + kk] = t[2];
      Bs[buf][(nn + 3) * LDB + kk] = t[3];
    }
  };

  v8f acc[2][4] = {};
  constexpr int kSteps = kF / 32;

  stage_a(0, 0);
  stage_b(0, 0);
  async_wait0();
  __syncthreads();

  for (int it = 0; it < kSteps; it++) {
    int cur = it & 1, nxt = cur ^ 1;
    if (it + 1 < kSteps) {
      stage_a(nxt, (it + 1) * 32);
      stage_b(nxt, (it + 1) * 32);
    }
    if (it + 2 < kSteps)  // prefetch the transpose-staged operand 2 ahead
      __builtin_prefetch(W + (size_t)((it + 2) * 32 + (tid >> 5)) * kF + n0, 0, 1);

    v16bf aF[2], bF[4];
#pragma unroll
    for (int i = 0; i < 2; i++)
      aF[i] = load_a_frag(&As[cur][(wm + i * 16) * LDA], LDA, l15, hh);
#pragma unroll
    for (int j = 0; j < 4; j++)
      bF[j] = load_b_frag(&Bs[cur][0], LDB, wn + j * 16 + l15, 0, hh);
#pragma unroll
    for (int i = 0; i < 2; i++)
#pragma unroll
      for (int j = 0; j < 4; j++)
        acc[i][j] = wmma_bf16(aF[i], bF[j], acc[i][j]);

    async_wait0();   // next tile fully in LDS (ASYNCcnt)
    __syncthreads(); // everyone done with cur, nxt writes visible
  }

  // Epilogue. C layout: VGPR r -> row r (lanes 0-15) / row 8+r (lanes 16-31).
#pragma unroll
  for (int i = 0; i < 2; i++) {
#pragma unroll
    for (int j = 0; j < 4; j++) {
      int col = n0 + wn + j * 16 + l15;
      float bcol = bias[col];
#pragma unroll
      for (int r = 0; r < 8; r++) {
        int row = m0 + wm + i * 16 + r + hh * 8;
        float v = acc[i][j][r] + bcol;
        if (args.qkv) {
          int h = col >> 6, d = col & 63;
          if (sel == 0)
            args.dq[((size_t)h * kT + row) * kD + d] = (bf16_t)v;
          else if (sel == 1)
            args.dk[((size_t)h * kTotal + (kOffset + row)) * kD + d] = (bf16_t)v;
          else
            args.dv[((size_t)h * kTotal + (kOffset + row)) * kD + d] = (bf16_t)v;
        } else {
          args.fout[(size_t)row * kF + col] = v;
        }
      }
    }
  }
}

// ---------------------------------------------------------------------------
// Kernel 4: flash attention. Block = (head, 128 q-rows), 8 waves x 16 rows.
// Double-buffered 64-position KV tiles (async K staging, VGPR-transposed V);
// online softmax; P transits per-wave LDS scratch for C->A layout conversion
// (same-wave LDS ops are in-order, no barrier needed for that).
// ---------------------------------------------------------------------------
__global__ void __launch_bounds__(256)
attn_wmma_kernel(const bf16_t* __restrict__ Q, const bf16_t* __restrict__ Kf,
                 const bf16_t* __restrict__ Vf, bf16_t* __restrict__ Cb) {
  constexpr int LDK = 72, LDP = 72;
  __shared__ alignas(16) bf16_t Ks[2][64 * LDK];   // [pos][d]
  __shared__ alignas(16) bf16_t Vs[2][64 * LDK];   // transposed [d][pos]
  __shared__ alignas(16) bf16_t Ps[8 * 16 * LDP];  // per-wave P scratch

  int h    = blockIdx.x;
  int tid  = threadIdx.x;
  int wave = tid >> 5, lane = tid & 31;
  int hh = lane >> 4, l15 = lane & 15;
  int qrow = blockIdx.y * 128 + wave * 16;

  const bf16_t* kbase0 = Kf + (size_t)h * kTotal * kD;
  const bf16_t* vbase0 = Vf + (size_t)h * kTotal * kD;

  auto stage_kv = [&](int buf, int kv0) {
    const bf16_t* kg = kbase0 + (size_t)kv0 * kD;
    const bf16_t* vg = vbase0 + (size_t)kv0 * kD;
    // K tile: 64x64 straight copy, 16B async chunks (2 per thread).
#pragma unroll
    for (int i = 0; i < 2; i++) {
      int c = tid + 256 * i;
      int r = c >> 3, cc = (c & 7) * 8;
      async_copy16(&Ks[buf][r * LDK + cc], kg + r * kD + cc);
    }
    // V tile: transpose into [d][pos] (VGPR path, scatter stores).
#pragma unroll
    for (int i = 0; i < 4; i++) {
      int c = tid + 256 * i;
      int r = c >> 4, cc = (c & 15) * 4;
      v4bf t = *(const v4bf*)(vg + r * kD + cc);
      Vs[buf][(cc + 0) * LDK + r] = t[0];
      Vs[buf][(cc + 1) * LDK + r] = t[1];
      Vs[buf][(cc + 2) * LDK + r] = t[2];
      Vs[buf][(cc + 3) * LDK + r] = t[3];
    }
  };

  // Q fragments for the 16x64 q-tile (two 32-deep k-steps over head dim).
  const bf16_t* qbase = Q + ((size_t)h * kT + qrow) * kD;
  v16bf qF[2];
#pragma unroll
  for (int kb = 0; kb < 2; kb++)
    qF[kb] = load_a_frag(qbase + kb * 32, kD, l15, hh);

  float m[8], lsum[8];
#pragma unroll
  for (int r = 0; r < 8; r++) { m[r] = -1e30f; lsum[r] = 0.f; }
  v8f o[4] = {};

  bf16_t* pw = Ps + wave * 16 * LDP;
  constexpr int kvSteps = kTotal / 64;

  stage_kv(0, 0);
  async_wait0();
  __syncthreads();

  for (int it = 0; it < kvSteps; it++) {
    int cur = it & 1, nxt = cur ^ 1;
    if (it + 1 < kvSteps) stage_kv(nxt, (it + 1) * 64);
    if (it + 2 < kvSteps)  // prefetch the transpose-staged V tile 2 ahead
      __builtin_prefetch(vbase0 + (size_t)((it + 2) * 64 + (tid >> 2)) * kD, 0, 1);

    // S = Q x K^T (Ks already [n=pos][k=d] for B fragments).
    v8f s[4];
#pragma unroll
    for (int j = 0; j < 4; j++) {
      v8f z = {};
      v16bf b0 = load_b_frag(&Ks[cur][0], LDK, j * 16 + l15, 0, hh);
      v16bf b1 = load_b_frag(&Ks[cur][0], LDK, j * 16 + l15, 32, hh);
      s[j] = wmma_bf16(qF[0], b0, z);
      s[j] = wmma_bf16(qF[1], b1, s[j]);
    }

    // Online softmax; rows of one C-VGPR live in one 16-lane half.
#pragma unroll
    for (int r = 0; r < 8; r++) {
      float mx = -1e30f;
#pragma unroll
      for (int j = 0; j < 4; j++) {
        s[j][r] *= 0.125f;  // 1/sqrt(64)
        mx = fmaxf(mx, s[j][r]);
      }
#pragma unroll
      for (int w = 1; w < 16; w <<= 1) mx = fmaxf(mx, __shfl_xor(mx, w, 32));
      float mnew  = fmaxf(m[r], mx);
      float alpha = __expf(m[r] - mnew);
      m[r] = mnew;
      float rs = 0.f;
#pragma unroll
      for (int j = 0; j < 4; j++) {
        float p = __expf(s[j][r] - mnew);
        rs += p;
        pw[(r + hh * 8) * LDP + j * 16 + l15] = (bf16_t)p;  // C-layout store
      }
#pragma unroll
      for (int w = 1; w < 16; w <<= 1) rs += __shfl_xor(rs, w, 32);
      lsum[r] = lsum[r] * alpha + rs;
#pragma unroll
      for (int jd = 0; jd < 4; jd++) o[jd][r] *= alpha;
    }

    // Reload P in A-layout (same-wave LDS ops are ordered; no barrier needed).
    v16bf pF[2];
#pragma unroll
    for (int kb = 0; kb < 2; kb++)
      pF[kb] = load_a_frag(pw + kb * 32, LDP, l15, hh);

    // O += P x V (Vs is [n=d][k=pos] for B fragments).
#pragma unroll
    for (int jd = 0; jd < 4; jd++) {
      v16bf vb0 = load_b_frag(&Vs[cur][0], LDK, jd * 16 + l15, 0, hh);
      v16bf vb1 = load_b_frag(&Vs[cur][0], LDK, jd * 16 + l15, 32, hh);
      o[jd] = wmma_bf16(pF[0], vb0, o[jd]);
      o[jd] = wmma_bf16(pF[1], vb1, o[jd]);
    }

    async_wait0();   // next KV tile fully in LDS (ASYNCcnt)
    __syncthreads(); // everyone done with cur, nxt writes visible
  }

  // Normalize and write ctx as bf16 [t][h*64+d] for the output projection.
#pragma unroll
  for (int jd = 0; jd < 4; jd++) {
#pragma unroll
    for (int r = 0; r < 8; r++) {
      int t = qrow + r + hh * 8;
      int d = jd * 16 + l15;
      Cb[(size_t)t * kF + h * kD + d] = (bf16_t)(o[jd][r] / lsum[r]);
    }
  }
}

// ---------------------------------------------------------------------------
// Host launcher.
// ---------------------------------------------------------------------------
extern "C" void kernel_launch(void* const* d_in, const int* in_sizes, int n_in,
                              void* d_out, int out_size, void* d_ws,
                              size_t ws_size, hipStream_t stream) {
  (void)in_sizes; (void)n_in; (void)out_size; (void)ws_size;
  const float* x  = (const float*)d_in[0];
  const float* Wq = (const float*)d_in[1];
  const float* bq = (const float*)d_in[2];
  const float* Wk = (const float*)d_in[3];
  const float* bk = (const float*)d_in[4];
  const float* Wv = (const float*)d_in[5];
  const float* bv = (const float*)d_in[6];
  const float* Wo = (const float*)d_in[7];
  const float* bo = (const float*)d_in[8];
  const float* kc = (const float*)d_in[9];
  const float* vc = (const float*)d_in[10];
  const int*   bt = (const int*)d_in[11];
  float* out = (float*)d_out;

  char* ws = (char*)d_ws;
  size_t off = 0;
  auto carve = [&](size_t elems) {
    bf16_t* p = (bf16_t*)(ws + off);
    off += (elems * sizeof(bf16_t) + 255) & ~(size_t)255;
    return p;
  };
  bf16_t* xb    = carve((size_t)kT * kF);
  bf16_t* Wqb   = carve((size_t)kF * kF);
  bf16_t* Wkb   = carve((size_t)kF * kF);
  bf16_t* Wvb   = carve((size_t)kF * kF);
  bf16_t* Wob   = carve((size_t)kF * kF);
  bf16_t* Qb    = carve((size_t)kH * kT * kD);
  bf16_t* Kfull = carve((size_t)kH * kTotal * kD);
  bf16_t* Vfull = carve((size_t)kH * kTotal * kD);
  bf16_t* Cb    = carve((size_t)kT * kF);

  const int n4 = (kT * kF) / 4;  // 262144
  cvt_f32_bf16_kernel<<<n4 / 256, 256, 0, stream>>>(x, xb, n4);
  cvt_f32_bf16_kernel<<<n4 / 256, 256, 0, stream>>>(Wq, Wqb, n4);
  cvt_f32_bf16_kernel<<<n4 / 256, 256, 0, stream>>>(Wk, Wkb, n4);
  cvt_f32_bf16_kernel<<<n4 / 256, 256, 0, stream>>>(Wv, Wvb, n4);
  cvt_f32_bf16_kernel<<<n4 / 256, 256, 0, stream>>>(Wo, Wob, n4);

  gather_kv_kernel<<<(kOffset * kH * (kD / 4)) / 256, 256, 0, stream>>>(
      kc, vc, bt, Kfull, Vfull);

  GemmArgs g{};
  g.X = xb; g.W0 = Wqb; g.W1 = Wkb; g.W2 = Wvb;
  g.b0 = bq; g.b1 = bk; g.b2 = bv;
  g.dq = Qb; g.dk = Kfull; g.dv = Vfull;
  g.fout = nullptr; g.qkv = 1;
  gemm_wmma_kernel<<<dim3(8, 8, 3), 256, 0, stream>>>(g);

  attn_wmma_kernel<<<dim3(kH, kT / 128), 256, 0, stream>>>(Qb, Kfull, Vfull, Cb);

  GemmArgs go{};
  go.X = Cb; go.W0 = Wob; go.b0 = bo;
  go.fout = out; go.qkv = 0;
  gemm_wmma_kernel<<<dim3(8, 8, 1), 256, 0, stream>>>(go);
}
